// CayleyNet_46600395162210
// MI455X (gfx1250) — matrix operand order repaired and checked by reading, not compiled, not run on hardware
//
#include <hip/hip_runtime.h>
#include <hip/hip_bf16.h>
#include <stdint.h>
#include <math.h>

#define NNODES 20000
#define NEDGES 320000
#define HIDF   128
#define NOUTF  10
#define KPOOL  18000
#define NP     (NNODES * HIDF)
#define NTILES (NNODES / 16)        // 1250 row tiles

typedef __attribute__((ext_vector_type(16))) _Float16 v16h;
typedef __attribute__((ext_vector_type(8)))  float    v8f;

// ---------------- WMMA GEMM: C[n,o] (+)= scale * sum_k A[n,k]*W(o,k) ----------------
// A: [NNODES,128] row-major f32.  W element (o,k) at W[(o*128+k)*wstride].
// Block = 256 threads (8 waves). Whole 128x128 weight tile is converted to f16 and
// staged in LDS once per block; wave w computes row-tile (blockIdx*8+w) x all 128 cols:
// per K-step one A fragment feeds 8 WMMAs whose B fragments stream from LDS.
__global__ void gemm_wmma_kernel(const float* __restrict__ A,
                                 const float* __restrict__ W, int wstride,
                                 float scale, float* __restrict__ C,
                                 int accumulate)
{
    __shared__ __align__(32) _Float16 lw[HIDF * HIDF];   // 32 KB: lw[o*128 + k]

    // cooperative stage + f32->f16 convert of the weight tile
    for (int i = threadIdx.x; i < HIDF * HIDF; i += 256)
        lw[i] = (_Float16)W[(long)i * wstride];
    __syncthreads();

    int lane = threadIdx.x & 31;
    int wave = threadIdx.x >> 5;
    int hi   = lane >> 4;
    int l16  = lane & 15;
    int tile = blockIdx.x * 8 + wave;
    if (tile >= NTILES) return;
    int row0 = tile * 16;

    v8f c[8];
#pragma unroll
    for (int t = 0; t < 8; ++t) c[t] = (v8f){};

#pragma unroll
    for (int kb = 0; kb < 4; ++kb) {
        // A 16-bit 16x32 fragment: lanes0-15 K=0..15, lanes16-31 K=16..31
        const float* arow = A + (long)(row0 + l16) * HIDF + kb * 32;
        v16h a;
#pragma unroll
        for (int v = 0; v < 8; ++v) {
#pragma unroll
            for (int p = 0; p < 2; ++p) {
                int ka = 2 * v + p + (hi ? 16 : 0);
                a[2 * v + p] = (_Float16)arow[ka];
            }
        }
        // 8 column tiles: B fragment = 16 contiguous f16 per lane from LDS
#pragma unroll
        for (int t = 0; t < 8; ++t) {
            const v16h* bp = (const v16h*)&lw[(16 * t + l16) * HIDF + kb * 32 + (hi ? 16 : 0)];
            v16h b = *bp;
            c[t] = __builtin_amdgcn_wmma_f32_16x16x32_f16(false, a, false, b,
                                                          (short)0, c[t], false, false);
        }
    }
    // C layout: VGPR r -> (M=r, N=lane) lanes0-15 ; (M=8+r, N=lane-16) lanes16-31
#pragma unroll
    for (int t = 0; t < 8; ++t) {
#pragma unroll
        for (int r = 0; r < 8; ++r) {
            int m = row0 + r + (hi ? 8 : 0);
            int n = 16 * t + l16;
            float v = scale * c[t][r];
            if (accumulate) C[(long)m * HIDF + n] += v;
            else            C[(long)m * HIDF + n]  = v;
        }
    }
}

// ---------------- graph / pointwise kernels ----------------
__global__ void deg_kernel(const int* __restrict__ row, float* __restrict__ deg)
{
    int e = blockIdx.x * 256 + threadIdx.x;
    if (e < NEDGES) atomicAdd(&deg[row[e]], 1.0f);
}

__global__ void coeff_kernel(const float* __restrict__ deg,
                             const float* __restrict__ hbuf,
                             const float* __restrict__ abuf, int layer,
                             float* __restrict__ tl, float* __restrict__ bdia)
{
    int n = blockIdx.x * 256 + threadIdx.x;
    if (n >= NNODES) return;
    float h  = hbuf[layer];
    float al = abuf[layer];
    float a  = h * (deg[n] - al);         // h * l_dia
    float inv = 1.0f / (a * a + 1.0f);
    tl[2 * n]     = a * inv;              // Re 1/(a+i)
    tl[2 * n + 1] = -inv;                 // Im 1/(a+i)
    bdia[2 * n]     = (a * a - 1.0f) * inv;  // Re (a-i)/(a+i)
    bdia[2 * n + 1] = -2.0f * a * inv;       // Im (a-i)/(a+i)
}

__global__ void init_y_kernel(const float* __restrict__ src,
                              float* __restrict__ yr, float* __restrict__ yi,
                              int do_relu)
{
    int g = blockIdx.x * 256 + threadIdx.x;
    if (g >= NP) return;
    float v = src[g];
    if (do_relu) v = v > 0.0f ? v : 0.0f;
    yr[g] = v;
    yi[g] = 0.0f;
}

__global__ void relu_kernel(float* __restrict__ p)
{
    int g = blockIdx.x * 256 + threadIdx.x;
    if (g < NP) { float v = p[g]; p[g] = v > 0.0f ? v : 0.0f; }
}

__global__ void diag_kernel(const float* __restrict__ bdia,
                            const float* __restrict__ yr, const float* __restrict__ yi,
                            float* __restrict__ br, float* __restrict__ bi)
{
    int g = blockIdx.x * 256 + threadIdx.x;
    if (g >= NP) return;
    int n = g >> 7;
    float dr = bdia[2 * n], di = bdia[2 * n + 1];
    float vr = yr[g], vi = yi[g];
    br[g] = dr * vr - di * vi;
    bi[g] = dr * vi + di * vr;
}

__global__ void copy2_kernel(const float* __restrict__ sr, const float* __restrict__ si,
                             float* __restrict__ dr, float* __restrict__ di)
{
    int g = blockIdx.x * 256 + threadIdx.x;
    if (g < NP) { dr[g] = sr[g]; di[g] = si[g]; }
}

// complex edge SpMM: coeff = csign * h * tmp_left[row[e]]
// gather_row==0: gather col, scatter row   (B-multiply,  csign=-1)
// gather_row==1: gather row, scatter col   (Jacobi step, csign=+1)
__global__ void edge_spmm_kernel(const int* __restrict__ row, const int* __restrict__ col,
                                 const float* __restrict__ tl,
                                 const float* __restrict__ hbuf, int layer, float csign,
                                 const float* __restrict__ inr, const float* __restrict__ ini,
                                 float* __restrict__ outr, float* __restrict__ outi,
                                 int gather_row)
{
    long g = (long)blockIdx.x * 256 + threadIdx.x;
    if (g >= (long)NEDGES * HIDF) return;
    int e = (int)(g >> 7);
    int f = (int)(g & 127);
    int r = row[e], c = col[e];
    float h = hbuf[layer] * csign;
    float cr = h * tl[2 * r];
    float ci = h * tl[2 * r + 1];
    int gsrc = gather_row ? r : c;
    int gdst = gather_row ? c : r;
    float vr = inr[(long)gsrc * HIDF + f];
    float vi = ini[(long)gsrc * HIDF + f];
    atomicAdd(&outr[(long)gdst * HIDF + f], cr * vr - ci * vi);
    atomicAdd(&outi[(long)gdst * HIDF + f], cr * vi + ci * vr);
}

// ---------------- pooling / head ----------------
__global__ void norm_kernel(const float* __restrict__ poolw, float* __restrict__ invn)
{
    __shared__ float sh[128];
    float v = poolw[threadIdx.x];
    sh[threadIdx.x] = v * v;
    __syncthreads();
    for (int s = 64; s >= 1; s >>= 1) {
        if (threadIdx.x < s) sh[threadIdx.x] += sh[threadIdx.x + s];
        __syncthreads();
    }
    if (threadIdx.x == 0) invn[0] = rsqrtf(sh[0]);
}

__global__ void score_kernel(const float* __restrict__ x2, const float* __restrict__ poolw,
                             const float* __restrict__ invn,
                             float* __restrict__ score, uint32_t* __restrict__ keys)
{
    int wave = threadIdx.x >> 5, lane = threadIdx.x & 31;
    int n = blockIdx.x * 8 + wave;
    if (n >= NNODES) return;
    float s = 0.0f;
#pragma unroll
    for (int q = 0; q < 4; ++q) {
        int f = lane + 32 * q;
        s += x2[(long)n * HIDF + f] * poolw[f];
    }
    for (int m = 16; m >= 1; m >>= 1) s += __shfl_xor(s, m, 32);
    if (lane == 0) {
        float sc = tanhf(s * invn[0]);
        score[n] = sc;
        uint32_t b = __float_as_uint(sc);
        keys[n] = (b & 0x80000000u) ? ~b : (b | 0x80000000u); // ascending sortable
    }
}

__global__ void init_state_kernel(uint32_t* __restrict__ st)
{
    st[0] = 0u;          // prefix / threshold
    st[1] = KPOOL;       // remaining needed
    st[2] = 0u;          // tie ticket
}

__global__ void hist_kernel(const uint32_t* __restrict__ keys,
                            const uint32_t* __restrict__ st,
                            uint32_t* __restrict__ hist, int shift)
{
    __shared__ uint32_t h[256];
    h[threadIdx.x] = 0u;
    __syncthreads();
    int n = blockIdx.x * 256 + threadIdx.x;
    if (n < NNODES) {
        uint32_t k = keys[n];
        uint32_t mask = (shift >= 24) ? 0u : (0xFFFFFFFFu << (shift + 8));
        if ((k & mask) == (st[0] & mask))
            atomicAdd(&h[(k >> shift) & 255u], 1u);
    }
    __syncthreads();
    if (h[threadIdx.x]) atomicAdd(&hist[threadIdx.x], h[threadIdx.x]);
}

__global__ void scan_kernel(const uint32_t* __restrict__ hist,
                            uint32_t* __restrict__ st, int shift)
{
    if (threadIdx.x != 0 || blockIdx.x != 0) return;
    uint32_t rem = st[1], pref = st[0];
    for (int d = 255; d >= 0; --d) {
        uint32_t h = hist[d];
        if (h >= rem) { pref |= ((uint32_t)d) << shift; break; }
        rem -= h;
    }
    st[0] = pref;
    st[1] = rem;
}

__global__ void select_kernel(const uint32_t* __restrict__ keys,
                              const float* __restrict__ score,
                              uint32_t* __restrict__ st, float* __restrict__ selw)
{
    int n = blockIdx.x * 256 + threadIdx.x;
    if (n >= NNODES) return;
    uint32_t k = keys[n], T = st[0], rem = st[1];
    float w = 0.0f;
    if (k > T) w = score[n];
    else if (k == T) { if (atomicAdd(&st[2], 1u) < rem) w = score[n]; }
    selw[n] = w;
}

__global__ void pool_sum_kernel(const float* __restrict__ selw,
                                const float* __restrict__ x2,
                                float* __restrict__ pooled)
{
    int f  = threadIdx.x;             // 128 threads
    int n0 = blockIdx.x * 160;        // 125 blocks * 160 nodes = 20000
    float acc = 0.0f;
    for (int i = 0; i < 160; ++i) {
        int n = n0 + i;
        acc += selw[n] * x2[(long)n * HIDF + f];
    }
    atomicAdd(&pooled[f], acc);
}

__global__ void final_kernel(const float* __restrict__ pooled,
                             const float* __restrict__ linW,
                             const float* __restrict__ linb, float* __restrict__ out)
{
    int o = threadIdx.y, lane = threadIdx.x; // block (32,10)
    float s = 0.0f;
    for (int f = lane; f < HIDF; f += 32) s += pooled[f] * linW[o * HIDF + f];
    for (int m = 16; m >= 1; m >>= 1) s += __shfl_xor(s, m, 32);
    if (lane == 0) out[o] = linb[o] + s * (1.0f / (float)KPOOL);
}

// ---------------- host orchestration ----------------
extern "C" void kernel_launch(void* const* d_in, const int* in_sizes, int n_in,
                              void* d_out, int out_size, void* d_ws, size_t ws_size,
                              hipStream_t stream)
{
    const float* x     = (const float*)d_in[0];
    const int*   ei    = (const int*)d_in[1];
    const float* Wr1   = (const float*)d_in[2];
    const float* Wc1   = (const float*)d_in[3];
    const float* Wr2   = (const float*)d_in[4];
    const float* Wc2   = (const float*)d_in[5];
    const float* hbuf  = (const float*)d_in[6];
    const float* abuf  = (const float*)d_in[7];
    const float* poolw = (const float*)d_in[8];
    const float* linW  = (const float*)d_in[9];
    const float* linb  = (const float*)d_in[10];
    float* out = (float*)d_out;

    const int* row = ei;
    const int* col = ei + NEDGES;

    float* W   = (float*)d_ws;
    float* P0r = W;                 float* P0i = W + 1L * NP;
    float* P1r = W + 2L * NP;       float* P1i = W + 3L * NP;
    float* P2r = W + 4L * NP;       float* P2i = W + 5L * NP;
    float* ACC = W + 6L * NP;
    float* S   = W + 7L * NP;
    float* deg  = S;
    float* tl   = S + 1L * NNODES;   // [N,2]
    float* bdia = S + 3L * NNODES;   // [N,2]
    float* score = S + 5L * NNODES;
    float* selw  = S + 6L * NNODES;
    uint32_t* keys = (uint32_t*)(S + 7L * NNODES);
    float* pooled = S + 8L * NNODES;
    float* invn   = pooled + 128;
    uint32_t* hist = (uint32_t*)(invn + 1);
    uint32_t* st   = hist + 256;

    dim3 b256(256);
    int gN   = (NNODES + 255) / 256;
    int gNP  = (NP + 255) / 256;
    int gEF  = (int)(((long)NEDGES * HIDF + 255) / 256);
    int gGemm = (NTILES + 7) / 8;   // 8 row-tiles (waves) per block

    // degree (shared by both layers)
    hipMemsetAsync(deg, 0, NNODES * sizeof(float), stream);
    deg_kernel<<<(NEDGES + 255) / 256, b256, 0, stream>>>(row, deg);

    const float* Wreal[2] = {Wr1, Wr2};
    const float* Wcplx[2] = {Wc1, Wc2};

    for (int layer = 0; layer < 2; ++layer) {
        coeff_kernel<<<gN, b256, 0, stream>>>(deg, hbuf, abuf, layer, tl, bdia);
        const float* xin = (layer == 0) ? x : ACC;
        // y0 = (relu?)x ; also serves as A for the real-term GEMM
        init_y_kernel<<<gNP, b256, 0, stream>>>(xin, P0r, P0i, layer == 0 ? 0 : 1);
        gemm_wmma_kernel<<<gGemm, b256, 0, stream>>>(P0r, Wreal[layer], 1, 1.0f, ACC, 0);

        float *ar = P0r, *ai = P0i, *br = P2r, *bi = P2i;
        for (int j = 0; j < 2; ++j) {
            // b_j = B @ y : diag part then edge scatter (gather col -> row)
            diag_kernel<<<gNP, b256, 0, stream>>>(bdia, ar, ai, P1r, P1i);
            edge_spmm_kernel<<<gEF, b256, 0, stream>>>(row, col, tl, hbuf, layer, -1.0f,
                                                       ar, ai, P1r, P1i, 0);
            // K=3 Jacobi iterations: yk = Scatter(jacobi * yk_prev) + b_j
            copy2_kernel<<<gNP, b256, 0, stream>>>(P1r, P1i, br, bi);
            edge_spmm_kernel<<<gEF, b256, 0, stream>>>(row, col, tl, hbuf, layer, 1.0f,
                                                       P1r, P1i, br, bi, 1);
            copy2_kernel<<<gNP, b256, 0, stream>>>(P1r, P1i, ar, ai);
            edge_spmm_kernel<<<gEF, b256, 0, stream>>>(row, col, tl, hbuf, layer, 1.0f,
                                                       br, bi, ar, ai, 1);
            copy2_kernel<<<gNP, b256, 0, stream>>>(P1r, P1i, br, bi);
            edge_spmm_kernel<<<gEF, b256, 0, stream>>>(row, col, tl, hbuf, layer, 1.0f,
                                                       ar, ai, br, bi, 1);
            // out += 2*Re(y @ Wc^T) = 2*yr@WrT - 2*yi@WiT
            const float* wb = Wcplx[layer] + (long)j * HIDF * HIDF * 2;
            gemm_wmma_kernel<<<gGemm, b256, 0, stream>>>(br, wb + 0, 2,  2.0f, ACC, 1);
            gemm_wmma_kernel<<<gGemm, b256, 0, stream>>>(bi, wb + 1, 2, -2.0f, ACC, 1);
            // final yk lives in (br,bi): it is next j's y
            float* t;
            t = ar; ar = br; br = t;
            t = ai; ai = bi; bi = t;
        }
    }

    // x2 = relu(ACC) in place
    relu_kernel<<<gNP, b256, 0, stream>>>(ACC);

    // scores + sortable keys
    norm_kernel<<<1, 128, 0, stream>>>(poolw, invn);
    score_kernel<<<(NNODES + 7) / 8, b256, 0, stream>>>(ACC, poolw, invn, score, keys);

    // radix-select the 18000-th largest key
    init_state_kernel<<<1, 1, 0, stream>>>(st);
    const int shifts[4] = {24, 16, 8, 0};
    for (int rnd = 0; rnd < 4; ++rnd) {
        hipMemsetAsync(hist, 0, 256 * sizeof(uint32_t), stream);
        hist_kernel<<<gN, b256, 0, stream>>>(keys, st, hist, shifts[rnd]);
        scan_kernel<<<1, 1, 0, stream>>>(hist, st, shifts[rnd]);
    }
    select_kernel<<<gN, b256, 0, stream>>>(keys, score, st, selw);

    // pooled[f] = sum_n selw[n] * x2[n,f] ; out = pooled/KPOOL @ linW^T + linb
    hipMemsetAsync(pooled, 0, 128 * sizeof(float), stream);
    pool_sum_kernel<<<125, 128, 0, stream>>>(selw, ACC, pooled);
    final_kernel<<<1, dim3(32, 10), 0, stream>>>(pooled, linW, linb, out);
}
